// ExpertBlock_27333171871857
// MI455X (gfx1250) — compile-verified, matrix operand order checked by reading, not run on hardware
//
#include <hip/hip_runtime.h>

#define T_TOK 2048
#define H_DIM 2048
#define I_DIM 1792
#define E_NUM 16
#define K_TOP 4
#define BM 128
#define BN 64
#define BK 32
#define MAXROWS (T_TOK * K_TOP + E_NUM * BM) // 10240 padded rows max

typedef __attribute__((ext_vector_type(16))) __bf16 v16bf;
typedef __attribute__((ext_vector_type(8)))  float  v8f;
typedef __attribute__((ext_vector_type(4)))  unsigned int v4u;
typedef __attribute__((ext_vector_type(8)))  int v8i;
typedef __attribute__((ext_vector_type(4)))  int v4i;

#if defined(__HIP_DEVICE_COMPILE__) && __has_builtin(__builtin_amdgcn_tensor_load_to_lds)
#define HAVE_TDM 1
#else
#define HAVE_TDM 0
#endif

// ---- fp32 -> bf16 conversion helpers ----
// gfx1250 has v_cvt_pk_bf16_f32 (observed in compiler output); emit the packed
// form directly: 1 VALU op per dword instead of 2 converts + shift/or repack.
#if defined(__HIP_DEVICE_COMPILE__)
__device__ __forceinline__ unsigned int f2bf2(float a, float b) {
  unsigned int r;
  asm("v_cvt_pk_bf16_f32 %0, %1, %2" : "=v"(r) : "v"(a), "v"(b));
  return r;  // src0 -> [15:0], src1 -> [31:16]
}
__device__ __forceinline__ unsigned short f2bf(float f) {
  unsigned int r;
  asm("v_cvt_pk_bf16_f32 %0, %1, 0" : "=v"(r) : "v"(f));
  return (unsigned short)r;
}
#else
// host pass only (never executed on device)
__device__ __forceinline__ unsigned short f2bf(float f) {
  unsigned int u = __builtin_bit_cast(unsigned int, f);
  u += 0x7FFFu + ((u >> 16) & 1u);
  return (unsigned short)(u >> 16);
}
__device__ __forceinline__ unsigned int f2bf2(float a, float b) {
  return (unsigned int)f2bf(a) | ((unsigned int)f2bf(b) << 16);
}
#endif

// LDS tile: row-major halfwords, row stride 34 halfwords (17 dwords) to kill
// bank conflicts. Fragment layout per ISA 16-bit A/B: lane<16 -> K base 0,
// lane>=16 -> K base 8; VGPR v<4 holds K pair (kb+2v), v>=4 holds (16+kb+2v).
__device__ __forceinline__ v16bf ld_frag(const unsigned int* lds, int row, int kb2) {
  int b = row * 17 + kb2;
  union { unsigned int u[8]; v16bf v; } f;
#pragma unroll
  for (int i = 0; i < 4; ++i) { f.u[i] = lds[b + i]; f.u[i + 4] = lds[b + 8 + i]; }
  return f.v;
}

#if HAVE_TDM
// One 2D TDM tile load: BM x BK bf16 tile, tensor row stride I_DIM elems,
// LDS pad 1 dword per 16 dwords -> native 68B row stride (conflict-free).
__device__ __forceinline__ void tdm_load_tile(unsigned int lds_addr, unsigned long long ga) {
  v4u g0;
  g0[0] = 1u;                                           // count=1, user mode
  g0[1] = lds_addr;                                     // lds_addr
  g0[2] = (unsigned int)ga;                             // global_addr[31:0]
  g0[3] = (unsigned int)((ga >> 32) & 0x01FFFFFFu) | 0x80000000u; // addr[56:32] | type=2
  v8i g1;
  g1[0] = (int)((1u << 16) | (1u << 20) | (3u << 22));  // 2B elems, pad_en, 1dw/16dw
  g1[1] = (int)(((unsigned)I_DIM & 0xFFFFu) << 16);     // tensor_dim0[15:0]
  g1[2] = (int)((((unsigned)I_DIM >> 16) & 0xFFFFu) |
                (((unsigned)MAXROWS & 0xFFFFu) << 16)); // dim0 hi | tensor_dim1 lo
  g1[3] = (int)(32u << 16);                             // tile_dim0 = 32 elems
  g1[4] = BM;                                           // tile_dim1 = 128 rows
  g1[5] = I_DIM;                                        // tensor_dim0_stride lo
  g1[6] = 0;
  g1[7] = 0;
  v4i gz = {0, 0, 0, 0};
#if __has_include(<hip/amd_detail/amd_gfx1250_TDM.h>)
  v8i gz8 = {};
  __builtin_amdgcn_tensor_load_to_lds(g0, g1, gz, gz, gz8, 0);
#else
  __builtin_amdgcn_tensor_load_to_lds(g0, g1, gz, gz, 0);
#endif
}
#endif

// ---------------- init: zero out/counters, poison row map ----------------
__global__ __launch_bounds__(256) void moe_init(float* out, int* cnt, int* row_tok) {
  int i = blockIdx.x * 256 + threadIdx.x;
  if (i < T_TOK * H_DIM) out[i] = 0.f;
  if (i < E_NUM) cnt[i] = 0;
  if (i < MAXROWS) row_tok[i] = -1;
}

// ---------------- router: logits, top-4, slot assignment ----------------
__global__ __launch_bounds__(256) void moe_router(
    const float* __restrict__ x, const float* __restrict__ rw,
    const float* __restrict__ bias, int* __restrict__ cnt,
    int* __restrict__ a_e, int* __restrict__ a_s, float* __restrict__ a_w) {
  __shared__ float red[E_NUM][256];
  const int t = blockIdx.x, tid = threadIdx.x;
  float part[E_NUM];
#pragma unroll
  for (int e = 0; e < E_NUM; ++e) part[e] = 0.f;
  for (int h = tid; h < H_DIM; h += 256) {
    float xv = x[(size_t)t * H_DIM + h];
#pragma unroll
    for (int e = 0; e < E_NUM; ++e) part[e] += xv * rw[e * H_DIM + h];
  }
#pragma unroll
  for (int e = 0; e < E_NUM; ++e) red[e][tid] = part[e];
  __syncthreads();
  for (int s = 128; s > 0; s >>= 1) {
    if (tid < s) {
#pragma unroll
      for (int e = 0; e < E_NUM; ++e) red[e][tid] += red[e][tid + s];
    }
    __syncthreads();
  }
  if (tid == 0) {
    float lg[E_NUM], mx = -1e30f;
#pragma unroll
    for (int e = 0; e < E_NUM; ++e) { lg[e] = red[e][0] + bias[e]; mx = fmaxf(mx, lg[e]); }
    float ex[E_NUM];
#pragma unroll
    for (int e = 0; e < E_NUM; ++e) ex[e] = __expf(lg[e] - mx);
    int used = 0, idx[K_TOP]; float wv[K_TOP], wsum = 0.f;
    for (int k = 0; k < K_TOP; ++k) {
      int best = 0; float bv = -1.f;
      for (int e = 0; e < E_NUM; ++e)
        if (!((used >> e) & 1) && ex[e] > bv) { bv = ex[e]; best = e; }
      used |= 1 << best; idx[k] = best; wv[k] = bv; wsum += bv;
    }
    float inv = 1.f / wsum;
    for (int k = 0; k < K_TOP; ++k) {
      int slot = atomicAdd(&cnt[idx[k]], 1);
      a_e[t * K_TOP + k] = idx[k];
      a_s[t * K_TOP + k] = slot;
      a_w[t * K_TOP + k] = wv[k] * inv;
    }
  }
}

// ---------------- prefix sum with BM padding ----------------
__global__ void moe_prefix(const int* cnt, int* off) {
  if (threadIdx.x == 0 && blockIdx.x == 0) {
    int acc = 0;
    for (int e = 0; e < E_NUM; ++e) {
      off[e] = acc;
      acc += (cnt[e] + BM - 1) & ~(BM - 1);
    }
    off[E_NUM] = acc;
  }
}

// ---------------- scatter token/weight into padded rows ----------------
__global__ __launch_bounds__(256) void moe_build(
    const int* __restrict__ a_e, const int* __restrict__ a_s,
    const float* __restrict__ a_w, const int* __restrict__ off,
    int* __restrict__ row_tok, float* __restrict__ row_w) {
  int i = blockIdx.x * 256 + threadIdx.x;
  if (i >= T_TOK * K_TOP) return;
  int r = off[a_e[i]] + a_s[i];
  row_tok[r] = i >> 2;
  row_w[r] = a_w[i];
}

// ---------------- grouped GEMM: act = silu(x*Wg) * (x*Wu), bf16 out ----------------
// Padding rows are clamped to token 0: GEMM output row m depends only on A row m,
// and padded output rows are never scattered, so garbage there is harmless.
__global__ __launch_bounds__(256) void moe_gateup(
    const float* __restrict__ x, const float* __restrict__ gate,
    const float* __restrict__ up, const int* __restrict__ off,
    const int* __restrict__ row_tok, unsigned short* __restrict__ act) {
  __shared__ unsigned int ldsA[BM * 17];
  __shared__ unsigned int ldsBg[BN * 17];
  __shared__ unsigned int ldsBu[BN * 17];

  const int e = blockIdx.z;
  const int rstart = off[e];
  const int rows = off[e + 1] - rstart;
  const int rowbase = blockIdx.y * BM;
  if (rowbase >= rows) return;
  const int n0 = blockIdx.x * BN;
  const int tid = threadIdx.x;
  const int lane = tid & 31, wid = tid >> 5;
  const int wrow = wid >> 1, wcol = wid & 1;
  const int lrow = lane & 15, lhalf = lane >> 4;
  const int kb2 = lhalf * 4;
  const size_t gbase = (size_t)e * H_DIM * I_DIM;

  // this thread's 4 A-tile items (fixed across the K loop)
  int arow[4], acol[4]; const float* aptr[4];
#pragma unroll
  for (int n = 0; n < 4; ++n) {
    int it = tid + 256 * n;                 // 1024 items = 128 rows x 8 col-groups
    arow[n] = it >> 3;
    acol[n] = (it & 7) * 4;
    int tok = row_tok[rstart + rowbase + arow[n]];
    aptr[n] = x + (size_t)(tok < 0 ? 0 : tok) * H_DIM + acol[n];
  }
  // B staging: this thread handles a K-pair x 4 N values (packed dword stores)
  const int kp = tid >> 4;                  // 0..15 -> k = 2*kp
  const int nn = (tid & 15) * 4;
  const float* gptr = gate + gbase + (size_t)(2 * kp) * I_DIM + n0 + nn;
  const float* uptr = up   + gbase + (size_t)(2 * kp) * I_DIM + n0 + nn;

  // software pipeline: preload slab 0 into registers
  float4 pa[4], pga, pgb, pua, pub;
#pragma unroll
  for (int n = 0; n < 4; ++n) pa[n] = *(const float4*)(aptr[n]);
  pga = *(const float4*)(gptr);
  pgb = *(const float4*)(gptr + I_DIM);
  pua = *(const float4*)(uptr);
  pub = *(const float4*)(uptr + I_DIM);

  v8f accG[2][2] = {};
  v8f accU[2][2] = {};

  for (int k0 = 0; k0 < H_DIM; k0 += BK) {
    // commit staged registers to LDS (fp32 -> packed bf16)
#pragma unroll
    for (int n = 0; n < 4; ++n) {
      unsigned int* d = &ldsA[arow[n] * 17 + (acol[n] >> 1)];
      d[0] = f2bf2(pa[n].x, pa[n].y);
      d[1] = f2bf2(pa[n].z, pa[n].w);
    }
#pragma unroll
    for (int d = 0; d < 4; ++d) {
      ldsBg[(nn + d) * 17 + kp] = f2bf2((&pga.x)[d], (&pgb.x)[d]);
      ldsBu[(nn + d) * 17 + kp] = f2bf2((&pua.x)[d], (&pub.x)[d]);
    }
    // issue next slab's global loads (in flight during the WMMA block)
    const int kn = k0 + BK;
    if (kn < H_DIM) {
#pragma unroll
      for (int n = 0; n < 4; ++n) pa[n] = *(const float4*)(aptr[n] + kn);
      pga = *(const float4*)(gptr + (size_t)kn * I_DIM);
      pgb = *(const float4*)(gptr + (size_t)kn * I_DIM + I_DIM);
      pua = *(const float4*)(uptr + (size_t)kn * I_DIM);
      pub = *(const float4*)(uptr + (size_t)kn * I_DIM + I_DIM);
    }
    __syncthreads();

    v16bf aF0 = ld_frag(ldsA, wrow * 32 + lrow, kb2);
    v16bf aF1 = ld_frag(ldsA, wrow * 32 + 16 + lrow, kb2);
#pragma unroll
    for (int j = 0; j < 2; ++j) {
      v16bf bg = ld_frag(ldsBg, wcol * 32 + j * 16 + lrow, kb2);
      v16bf bu = ld_frag(ldsBu, wcol * 32 + j * 16 + lrow, kb2);
      accG[0][j] = __builtin_amdgcn_wmma_f32_16x16x32_bf16(false, aF0, false, bg, (short)0, accG[0][j], false, false);
      accG[1][j] = __builtin_amdgcn_wmma_f32_16x16x32_bf16(false, aF1, false, bg, (short)0, accG[1][j], false, false);
      accU[0][j] = __builtin_amdgcn_wmma_f32_16x16x32_bf16(false, aF0, false, bu, (short)0, accU[0][j], false, false);
      accU[1][j] = __builtin_amdgcn_wmma_f32_16x16x32_bf16(false, aF1, false, bu, (short)0, accU[1][j], false, false);
    }
    __syncthreads();
  }

  // fused SiLU epilogue, bf16 store
#pragma unroll
  for (int i = 0; i < 2; ++i) {
#pragma unroll
    for (int j = 0; j < 2; ++j) {
#pragma unroll
      for (int v = 0; v < 8; ++v) {
        int m = wrow * 32 + i * 16 + lhalf * 8 + v;   // C layout: M = v + 8*lhalf
        int col = n0 + wcol * 32 + j * 16 + lrow;     // N = lane%16
        float g = accG[i][j][v], u = accU[i][j][v];
        float a = (g / (1.f + __expf(-g))) * u;
        act[(size_t)(rstart + rowbase + m) * I_DIM + col] = f2bf(a);
      }
    }
  }
}

// ---------------- grouped GEMM: out += w * (act * Wd) ----------------
__global__ __launch_bounds__(256) void moe_down(
    const unsigned short* __restrict__ act, const float* __restrict__ dwn,
    const int* __restrict__ off, const int* __restrict__ row_tok,
    const float* __restrict__ row_w, float* __restrict__ out) {
  __shared__ unsigned int ldsA[2][BM * 17];   // double-buffered TDM destination
  __shared__ unsigned int ldsB[BN * 17];

  const int e = blockIdx.z;
  const int rstart = off[e];
  const int rows = off[e + 1] - rstart;
  const int rowbase = blockIdx.y * BM;
  if (rowbase >= rows) return;
  const int n0 = blockIdx.x * BN;
  const int tid = threadIdx.x;
  const int lane = tid & 31, wid = tid >> 5;
  const int wrow = wid >> 1, wcol = wid & 1;
  const int lrow = lane & 15, lhalf = lane >> 4;
  const int kb2 = lhalf * 4;
  const size_t dbase = (size_t)e * I_DIM * H_DIM;
  const int kp = tid >> 4;                  // B staging K-pair
  const int nn = (tid & 15) * 4;

#if HAVE_TDM
  const unsigned long long act_base =
      (unsigned long long)(const void*)act +
      ((unsigned long long)(rstart + rowbase) * I_DIM) * 2ull;
  const unsigned int lds_a0 = (unsigned int)(unsigned long long)(void*)&ldsA[0][0];
  const unsigned int lds_a1 = (unsigned int)(unsigned long long)(void*)&ldsA[1][0];
  if (tid < 32) tdm_load_tile(lds_a0, act_base);        // prime buffer 0
#endif

  // software pipeline B: preload slab 0
  const float* dptr = dwn + dbase + (size_t)(2 * kp) * H_DIM + n0 + nn;
  float4 pba = *(const float4*)(dptr);
  float4 pbb = *(const float4*)(dptr + H_DIM);

  v8f acc[2][2] = {};
  int cur = 0;

  for (int k0 = 0; k0 < I_DIM; k0 += BK, cur ^= 1) {
#if HAVE_TDM
    // issue next A tile into the other buffer (overlaps with this tile's WMMA)
    if (k0 + BK < I_DIM && tid < 32)
      tdm_load_tile(cur ? lds_a0 : lds_a1,
                    act_base + (unsigned long long)(k0 + BK) * 2ull);
#else
    // fallback A staging: bf16 act copy via VGPRs
#pragma unroll
    for (int n = 0; n < 2; ++n) {
      int it = tid + 256 * n;               // 512 items = 128 rows x 4 groups of 8
      int r = it >> 2, c = (it & 3) * 8;
      uint4 v = *(const uint4*)(act + (size_t)(rstart + rowbase + r) * I_DIM + k0 + c);
      unsigned int* d = &ldsA[cur][r * 17 + (c >> 1)];
      d[0] = v.x; d[1] = v.y; d[2] = v.z; d[3] = v.w;
    }
#endif
    // commit staged B regs, then issue next slab's B loads
#pragma unroll
    for (int d = 0; d < 4; ++d)
      ldsB[(nn + d) * 17 + kp] = f2bf2((&pba.x)[d], (&pbb.x)[d]);
    const int kn = k0 + BK;
    if (kn < I_DIM) {
      pba = *(const float4*)(dptr + (size_t)kn * H_DIM);
      pbb = *(const float4*)(dptr + (size_t)kn * H_DIM + H_DIM);
    }
#if HAVE_TDM
    // in-order TDM completion: 1 outstanding (the prefetched next tile) is OK
    if (k0 + BK < I_DIM) __builtin_amdgcn_s_wait_tensorcnt(1);
    else                 __builtin_amdgcn_s_wait_tensorcnt(0);
#endif
    __syncthreads();

    const unsigned int* lA = &ldsA[cur][0];
    v16bf aF0 = ld_frag(lA, wrow * 32 + lrow, kb2);
    v16bf aF1 = ld_frag(lA, wrow * 32 + 16 + lrow, kb2);
#pragma unroll
    for (int j = 0; j < 2; ++j) {
      v16bf bf = ld_frag(ldsB, wcol * 32 + j * 16 + lrow, kb2);
      acc[0][j] = __builtin_amdgcn_wmma_f32_16x16x32_bf16(false, aF0, false, bf, (short)0, acc[0][j], false, false);
      acc[1][j] = __builtin_amdgcn_wmma_f32_16x16x32_bf16(false, aF1, false, bf, (short)0, acc[1][j], false, false);
    }
    __syncthreads();
  }

  // weighted scatter-add into the token's output row (skips padding rows)
#pragma unroll
  for (int i = 0; i < 2; ++i) {
#pragma unroll
    for (int v = 0; v < 8; ++v) {
      int m = wrow * 32 + i * 16 + lhalf * 8 + v;
      int grow = rstart + rowbase + m;
      int tok = row_tok[grow];
      if (tok < 0) continue;
      float w = row_w[grow];
#pragma unroll
      for (int j = 0; j < 2; ++j) {
        int col = n0 + wcol * 32 + j * 16 + lrow;
        atomicAdd(out + (size_t)tok * H_DIM + col, w * acc[i][j][v]);
      }
    }
  }
}

extern "C" void kernel_launch(void* const* d_in, const int* in_sizes, int n_in,
                              void* d_out, int out_size, void* d_ws, size_t ws_size,
                              hipStream_t stream) {
  (void)in_sizes; (void)n_in; (void)out_size; (void)ws_size;
  const float* x    = (const float*)d_in[0];
  const float* rw   = (const float*)d_in[1];
  const float* bias = (const float*)d_in[2];
  const float* gate = (const float*)d_in[3];
  const float* up   = (const float*)d_in[4];
  const float* dwn  = (const float*)d_in[5];
  float* out = (float*)d_out;

  char* ws = (char*)d_ws;
  int*   cnt     = (int*)(ws + 0);
  int*   off     = (int*)(ws + 64);
  int*   a_e     = (int*)(ws + 256);
  int*   a_s     = (int*)(ws + 256 + 32768);
  float* a_w     = (float*)(ws + 256 + 65536);
  int*   row_tok = (int*)(ws + 256 + 98304);
  float* row_w   = (float*)(ws + 256 + 98304 + 40960);
  unsigned short* act = (unsigned short*)(ws + 180736); // 16B aligned; ~36.7 MB

  moe_init<<<dim3((T_TOK * H_DIM + 255) / 256), dim3(256), 0, stream>>>(out, cnt, row_tok);
  moe_router<<<dim3(T_TOK), dim3(256), 0, stream>>>(x, rw, bias, cnt, a_e, a_s, a_w);
  moe_prefix<<<dim3(1), dim3(1), 0, stream>>>(cnt, off);
  moe_build<<<dim3((T_TOK * K_TOP + 255) / 256), dim3(256), 0, stream>>>(a_e, a_s, a_w, off, row_tok, row_w);
  moe_gateup<<<dim3(I_DIM / BN, T_TOK / BM, E_NUM), dim3(256), 0, stream>>>(x, gate, up, off, row_tok, act);
  moe_down<<<dim3(H_DIM / BN, T_TOK / BM, E_NUM), dim3(256), 0, stream>>>(act, dwn, off, row_tok, row_w, out);
}
